// ESEGAT_65859028517063
// MI455X (gfx1250) — matrix operand order repaired and checked by reading, not compiled, not run on hardware
//
#include <hip/hip_runtime.h>

typedef unsigned short u16;
typedef __attribute__((ext_vector_type(16))) __bf16 v16bf;
typedef __attribute__((ext_vector_type(2)))  __bf16 v2bf;
typedef __attribute__((ext_vector_type(2)))  float  v2f;
typedef __attribute__((ext_vector_type(8)))  float  v8f;

#define NSN   100000
#define NEN   100000
#define EDGN  1600000
#define DIM   256
#define HEADS 8
#define HDN   32
#define FEATN 64
#define DFFN  512

// ---------- helpers ----------
__device__ inline u16 f2bf(float x) {
    union { __bf16 h; u16 u; } c; c.h = (__bf16)x; return c.u;
}
// packed 2xf32 -> 2xbf16 (selects v_cvt_pk_bf16_f32 with both operands)
__device__ inline unsigned pack2bf(float lo, float hi) {
    v2f f; f.x = lo; f.y = hi;
    union { v2bf v; unsigned u; } c;
    c.v = __builtin_convertvector(f, v2bf);
    return c.u;
}
__device__ inline float bflo(unsigned u) { union { unsigned u; float f; } c; c.u = u << 16;          return c.f; }
__device__ inline float bfhi(unsigned u) { union { unsigned u; float f; } c; c.u = u & 0xffff0000u;  return c.f; }

__device__ inline float gelu_tanh(float x) {
    float x3 = x * x * x;
    return 0.5f * x * (1.0f + tanhf(0.7978845608028654f * (x + 0.044715f * x3)));
}
__device__ inline void atomicMaxF(float* a, float v) {
    if (v >= 0.0f) atomicMax((int*)a, __float_as_int(v));
    else           atomicMin((unsigned int*)a, __float_as_uint(v));
}

// ---------- double-buffer staging regs (16 elements / thread) ----------
template <typename T> struct ARegs;
template <> struct ARegs<float> { float4 r[4]; };
template <> struct ARegs<u16>   { uint4  r[2]; };

__device__ inline void fetchA(ARegs<float>& a, const float* p, bool ok) {
    if (ok) {
        const float4* s4 = (const float4*)p;
        a.r[0] = s4[0]; a.r[1] = s4[1]; a.r[2] = s4[2]; a.r[3] = s4[3];
    } else {
        float4 z = {0.f, 0.f, 0.f, 0.f};
        a.r[0] = z; a.r[1] = z; a.r[2] = z; a.r[3] = z;
    }
}
__device__ inline void fetchA(ARegs<u16>& a, const u16* p, bool ok) {
    if (ok) { a.r[0] = ((const uint4*)p)[0]; a.r[1] = ((const uint4*)p)[1]; }
    else    { uint4 z = {0u,0u,0u,0u}; a.r[0] = z; a.r[1] = z; }
}
__device__ inline void storeA(u16* dst, const ARegs<float>& a) {
    uint4 o0, o1;
    o0.x = pack2bf(a.r[0].x, a.r[0].y); o0.y = pack2bf(a.r[0].z, a.r[0].w);
    o0.z = pack2bf(a.r[1].x, a.r[1].y); o0.w = pack2bf(a.r[1].z, a.r[1].w);
    o1.x = pack2bf(a.r[2].x, a.r[2].y); o1.y = pack2bf(a.r[2].z, a.r[2].w);
    o1.z = pack2bf(a.r[3].x, a.r[3].y); o1.w = pack2bf(a.r[3].z, a.r[3].w);
    *(uint4*)dst       = o0;
    *(uint4*)(dst + 8) = o1;
}
__device__ inline void storeA(u16* dst, const ARegs<u16>& a) {
    *(uint4*)dst       = a.r[0];
    *(uint4*)(dst + 8) = a.r[1];
}

// ---------- weight prep ----------
// fc_w[h,i,o] (f32) -> W[i, h*HD+o] (bf16)
__global__ void k_permw(const float* __restrict__ src, u16* __restrict__ dst, int K, int total) {
    int t = blockIdx.x * blockDim.x + threadIdx.x;
    if (t >= total) return;
    int o = t % HDN;
    int i = (t / HDN) % K;
    int h = t / (HDN * K);
    dst[(size_t)i * (HEADS * HDN) + h * HDN + o] = f2bf(src[t]);
}
__global__ void k_cvt(const float* __restrict__ src, u16* __restrict__ dst, int n) {
    int t = blockIdx.x * blockDim.x + threadIdx.x;
    if (t < n) dst[t] = f2bf(src[t]);
}
__global__ void k_init(float* __restrict__ hh, float* __restrict__ smax, float* __restrict__ den) {
    int t = blockIdx.x * blockDim.x + threadIdx.x;
    if (t < NSN * DIM) hh[t] = 0.0f;
    if (t < NSN * HEADS) { smax[t] = __int_as_float(0xff800000); den[t] = 0.0f; }
}

// ---------- WMMA GEMM: C[M,N] = A[M,K] * B[K,N], 64x64 block tile, K-step 32 ----------
// EPI: 0=f32, 1=bf16, 2=+bias f32, 3=+bias gelu bf16, 4=+bias +resid f32
template <typename AT, int EPI>
__launch_bounds__(128)
__global__ void k_gemm(const AT* __restrict__ A, const u16* __restrict__ B,
                       void* __restrict__ Cout, const float* __restrict__ bias,
                       const float* __restrict__ resid, int M, int K, int N) {
    __shared__ __align__(16) u16 As[64][40];   // 32 cols + pad
    __shared__ __align__(16) u16 Bs[32][72];   // 64 cols + pad
    const int t    = threadIdx.x;
    const int lane = t & 31;
    const int wv   = t >> 5;
    const int brow = blockIdx.y * 64;
    const int bcol = blockIdx.x * 64;

    v8f acc[4];
#pragma unroll
    for (int j = 0; j < 4; ++j) acc[j] = (v8f){0.f,0.f,0.f,0.f,0.f,0.f,0.f,0.f};

    union Frag { v16bf v; uint4 q[2]; };

    const int ar = t >> 1, ac = (t & 1) * 16;   // A stage: 64 rows x 32 cols, 16 elems/thread
    const int br = t >> 2, bc = (t & 3) * 16;   // B stage: 32 rows x 64 cols, 16 elems/thread
    const bool arow_ok = (brow + ar) < M;
    const AT*  aptr = A + (size_t)(brow + ar) * K + ac;
    const u16* bptr = B + (size_t)br * N + bcol + bc;

    // prologue fetch (double buffer in registers)
    ARegs<AT> areg;
    uint4 breg0, breg1;
    fetchA(areg, aptr, arow_ok);
    breg0 = ((const uint4*)bptr)[0];
    breg1 = ((const uint4*)(bptr + 8))[0];

    for (int k0 = 0; k0 < K; k0 += 32) {
        // commit staged tile to LDS
        storeA(&As[ar][ac], areg);
        *(uint4*)&Bs[br][bc]     = breg0;
        *(uint4*)&Bs[br][bc + 8] = breg1;
        __syncthreads();

        // issue next tile's global loads now; they complete during the WMMA phase
        if (k0 + 32 < K) {
            fetchA(areg, aptr + k0 + 32, arow_ok);
            const u16* bp = bptr + (size_t)(k0 + 32) * N;
            breg0 = ((const uint4*)bp)[0];
            breg1 = ((const uint4*)(bp + 8))[0];
        }

        // A fragment: 16x32 bf16, ISA layout: lane<16 -> K {kb..kb+7, 16+kb..}, kb=8*(lane>=16)
        Frag a;
        const int am = wv * 16 + (lane & 15);
        const int kb = (lane >> 4) * 8;
        a.q[0] = *(const uint4*)&As[am][kb];
        a.q[1] = *(const uint4*)&As[am][16 + kb];

#pragma unroll
        for (int j = 0; j < 4; ++j) {
            // B fragment: 32x16 bf16, col n = lane&15, K {kb..kb+7, 16+kb..}, kb=8*(lane>=16)
            Frag b;
            const int bn = j * 16 + (lane & 15);
#pragma unroll
            for (int s = 0; s < 8; ++s) {
                b.v[s]     = __builtin_bit_cast(__bf16, Bs[kb + s][bn]);
                b.v[8 + s] = __builtin_bit_cast(__bf16, Bs[16 + kb + s][bn]);
            }
            acc[j] = __builtin_amdgcn_wmma_f32_16x16x32_bf16(
                false, a.v, false, b.v, (short)0, acc[j], false, false);
        }
        __syncthreads();
    }

    // epilogue: C/D layout: VGPR r -> row r + 8*(lane>=16), col = lane&15
    const int rbase = brow + wv * 16 + ((lane >> 4) << 3);
    const int cl    = lane & 15;
#pragma unroll
    for (int j = 0; j < 4; ++j) {
        int gc = bcol + j * 16 + cl;
        float bv = 0.0f;
        if (EPI >= 2) bv = bias[gc];
#pragma unroll
        for (int r = 0; r < 8; ++r) {
            int gr = rbase + r;
            if (gr < M) {
                float v = acc[j][r] + bv;
                if (EPI == 3) v = gelu_tanh(v);
                if (EPI == 4) v += resid[(size_t)gr * N + gc];
                if (EPI == 1 || EPI == 3) ((u16*)Cout)[(size_t)gr * N + gc] = f2bf(v);
                else                      ((float*)Cout)[(size_t)gr * N + gc] = v;
            }
        }
    }
}

// ---------- edge phase (wave per edge; lane covers 8 channels = one head per 4 lanes) ----------
__launch_bounds__(256)
__global__ void k_score(const u16* __restrict__ z, const u16* __restrict__ df,
                        const int* __restrict__ esrc, const int* __restrict__ edst,
                        float* __restrict__ score, float* __restrict__ smax) {
    int e = (blockIdx.x * 256 + threadIdx.x) >> 5;
    if (e >= EDGN) return;
    int lane = threadIdx.x & 31;
    int src = esrc[e], dst = edst[e];
    uint4 qz = *(const uint4*)(z  + (size_t)src * DIM + lane * 8);
    uint4 qd = *(const uint4*)(df + (size_t)dst * DIM + lane * 8);
    float p = bflo(qz.x)*bflo(qd.x) + bfhi(qz.x)*bfhi(qd.x)
            + bflo(qz.y)*bflo(qd.y) + bfhi(qz.y)*bfhi(qd.y)
            + bflo(qz.z)*bflo(qd.z) + bfhi(qz.z)*bfhi(qd.z)
            + bflo(qz.w)*bflo(qd.w) + bfhi(qz.w)*bfhi(qd.w);
    p += __shfl_xor(p, 1, 32);
    p += __shfl_xor(p, 2, 32);
    if ((lane & 3) == 0) {
        int h = lane >> 2;
        score[(size_t)e * HEADS + h] = p;
        atomicMaxF(&smax[(size_t)dst * HEADS + h], p);
    }
}

__launch_bounds__(256)
__global__ void k_exden(const int* __restrict__ edst, float* __restrict__ score,
                        const float* __restrict__ smax, float* __restrict__ den) {
    int t = blockIdx.x * 256 + threadIdx.x;
    if (t >= EDGN * HEADS) return;
    int e = t >> 3, h = t & 7;
    int dst = edst[e];
    float ex = __expf(score[t] - smax[(size_t)dst * HEADS + h]);
    score[t] = ex;
    atomicAdd(&den[(size_t)dst * HEADS + h], ex);
}

__launch_bounds__(256)
__global__ void k_agg(const u16* __restrict__ z, const float* __restrict__ ex,
                      const float* __restrict__ den, const int* __restrict__ esrc,
                      const int* __restrict__ edst, float* __restrict__ hh) {
    int e = (blockIdx.x * 256 + threadIdx.x) >> 5;
    if (e >= EDGN) return;
    int lane = threadIdx.x & 31;
    int src = esrc[e], dst = edst[e];
    int h = lane >> 2;
    float dn = den[(size_t)dst * HEADS + h];
    dn = dn > 0.0f ? dn : 1.0f;
    float al = ex[(size_t)e * HEADS + h] / dn;
    uint4 qz = *(const uint4*)(z + (size_t)src * DIM + lane * 8);
    float* hp = hh + (size_t)dst * DIM + lane * 8;
    atomicAdd(hp + 0, al * bflo(qz.x)); atomicAdd(hp + 1, al * bfhi(qz.x));
    atomicAdd(hp + 2, al * bflo(qz.y)); atomicAdd(hp + 3, al * bfhi(qz.y));
    atomicAdd(hp + 4, al * bflo(qz.z)); atomicAdd(hp + 5, al * bfhi(qz.z));
    atomicAdd(hp + 6, al * bflo(qz.w)); atomicAdd(hp + 7, al * bfhi(qz.w));
}

// ---------- elu + concat(h, s) -> bf16 [NS, 512] ----------
__launch_bounds__(256)
__global__ void k_hcat(const float* __restrict__ hh, const float* __restrict__ s,
                       u16* __restrict__ hcat) {
    int t = blockIdx.x * 256 + threadIdx.x;
    if (t >= NSN * DIM) return;
    int row = t >> 8, c = t & 255;
    float x = hh[t];
    float el = x > 0.0f ? x : (__expf(x) - 1.0f);
    hcat[(size_t)row * (2 * DIM) + c]       = f2bf(el);
    hcat[(size_t)row * (2 * DIM) + DIM + c] = f2bf(s[t]);
}

// ---------- layernorm (wave per row) -> bf16 ----------
__launch_bounds__(256)
__global__ void k_ln(const float* __restrict__ pre, const float* __restrict__ g,
                     const float* __restrict__ b, u16* __restrict__ xn) {
    int row = (blockIdx.x * 256 + threadIdx.x) >> 5;
    if (row >= NSN) return;
    int lane = threadIdx.x & 31;
    const float* x = pre + (size_t)row * DIM + lane * 8;
    float v[8], s = 0.0f, sq = 0.0f;
#pragma unroll
    for (int j = 0; j < 8; ++j) { v[j] = x[j]; s += v[j]; sq += v[j] * v[j]; }
#pragma unroll
    for (int m = 1; m < 32; m <<= 1) { s += __shfl_xor(s, m, 32); sq += __shfl_xor(sq, m, 32); }
    float mu  = s * (1.0f / DIM);
    float var = sq * (1.0f / DIM) - mu * mu;
    float rs  = rsqrtf(var + 1e-6f);
#pragma unroll
    for (int j = 0; j < 8; ++j) {
        int c = lane * 8 + j;
        xn[(size_t)row * DIM + c] = f2bf((v[j] - mu) * rs * g[c] + b[c]);
    }
}

extern "C" void kernel_launch(void* const* d_in, const int* in_sizes, int n_in,
                              void* d_out, int out_size, void* d_ws, size_t ws_size,
                              hipStream_t stream) {
    const float* s    = (const float*)d_in[0];
    const float* e    = (const float*)d_in[1];
    const float* dstf = (const float*)d_in[2];
    const float* fcw  = (const float*)d_in[3];
    const float* dfw  = (const float*)d_in[4];
    const float* pw   = (const float*)d_in[5];
    const float* pb   = (const float*)d_in[6];
    const float* lng  = (const float*)d_in[7];
    const float* lnb  = (const float*)d_in[8];
    const float* w1   = (const float*)d_in[9];
    const float* b1   = (const float*)d_in[10];
    const float* w2   = (const float*)d_in[11];
    const float* b2   = (const float*)d_in[12];
    const int* esrc   = (const int*)d_in[13];
    const int* edst   = (const int*)d_in[14];
    float* out = (float*)d_out;

    size_t off = 0;
    auto carve = [&](size_t bytes) -> char* {
        char* p = (char*)d_ws + off;
        off += (bytes + 255) & ~(size_t)255;
        return p;
    };
    u16*   zb   = (u16*)  carve((size_t)NEN * DIM * 2);       // z bf16          (dead after agg)
    float* sc   = (float*)carve((size_t)EDGN * HEADS * 4);    // score/ex f32    (dead after agg)
    u16*   dfb  = (u16*)  carve((size_t)NSN * DIM * 2);       // df bf16         (dead after score)
    float* hh   = (float*)carve((size_t)NSN * DIM * 4);       // aggregate f32
    u16*   hcat = (u16*)  carve((size_t)NSN * 2 * DIM * 2);   // cat(elu(h), s) bf16
    float* smax = (float*)carve((size_t)NSN * HEADS * 4);
    float* den  = (float*)carve((size_t)NSN * HEADS * 4);
    u16* wz  = (u16*)carve((size_t)DIM * DIM * 2);
    u16* wdf = (u16*)carve((size_t)FEATN * DIM * 2);
    u16* wpj = (u16*)carve((size_t)2 * DIM * DIM * 2);
    u16* w1b = (u16*)carve((size_t)DIM * DFFN * 2);
    u16* w2b = (u16*)carve((size_t)DFFN * DIM * 2);
    u16* gbuf = (u16*)zb;    // alias: z+score region (102.4MB) reused for gelu hidden bf16
    u16* xnb  = dfb;         // alias: df region reused for layernorm output bf16

    // weight prep + init
    k_permw<<<(HEADS * DIM * HDN + 255) / 256, 256, 0, stream>>>(fcw, wz, DIM, HEADS * DIM * HDN);
    k_permw<<<(HEADS * FEATN * HDN + 255) / 256, 256, 0, stream>>>(dfw, wdf, FEATN, HEADS * FEATN * HDN);
    k_cvt<<<(2 * DIM * DIM + 255) / 256, 256, 0, stream>>>(pw, wpj, 2 * DIM * DIM);
    k_cvt<<<(DIM * DFFN + 255) / 256, 256, 0, stream>>>(w1, w1b, DIM * DFFN);
    k_cvt<<<(DFFN * DIM + 255) / 256, 256, 0, stream>>>(w2, w2b, DFFN * DIM);
    k_init<<<(NSN * DIM + 255) / 256, 256, 0, stream>>>(hh, smax, den);

    dim3 blk(128);
    // z = e @ Wz -> bf16 [NE, 256]
    k_gemm<float, 1><<<dim3(DIM / 64, (NEN + 63) / 64), blk, 0, stream>>>(e, wz, zb, nullptr, nullptr, NEN, DIM, DIM);
    // df = dst_feat @ Wdf -> bf16 [NS, 256]
    k_gemm<float, 1><<<dim3(DIM / 64, (NSN + 63) / 64), blk, 0, stream>>>(dstf, wdf, dfb, nullptr, nullptr, NSN, FEATN, DIM);
    // edge phase: scores + segment softmax + weighted aggregate
    k_score<<<EDGN / 8, 256, 0, stream>>>(zb, dfb, esrc, edst, sc, smax);
    k_exden<<<(EDGN * HEADS) / 256, 256, 0, stream>>>(edst, sc, smax, den);
    k_agg<<<EDGN / 8, 256, 0, stream>>>(zb, sc, den, esrc, edst, hh);
    // elu + concat
    k_hcat<<<(NSN * DIM) / 256, 256, 0, stream>>>(hh, s, hcat);
    // pre = hcat @ Wproj + pb -> d_out (f32)
    k_gemm<u16, 2><<<dim3(DIM / 64, (NSN + 63) / 64), blk, 0, stream>>>(hcat, wpj, out, pb, nullptr, NSN, 2 * DIM, DIM);
    // xn = LN(pre) -> bf16
    k_ln<<<(NSN + 7) / 8, 256, 0, stream>>>(out, lng, lnb, xnb);
    // g = gelu(xn @ W1 + b1) -> bf16 [NS, 512]
    k_gemm<u16, 3><<<dim3(DFFN / 64, (NSN + 63) / 64), blk, 0, stream>>>(xnb, w1b, gbuf, b1, nullptr, NSN, DIM, DFFN);
    // out = pre + (g @ W2 + b2)
    k_gemm<u16, 4><<<dim3(DIM / 64, (NSN + 63) / 64), blk, 0, stream>>>(gbuf, w2b, out, b2, out, NSN, DFFN, DIM);
}
